// TemporalGraphormerMultiHeadAttention_18021682774303
// MI455X (gfx1250) — matrix operand order repaired and checked by reading, not compiled, not run on hardware
//
#include <hip/hip_runtime.h>
#include <hip/hip_bf16.h>
#include <math.h>

#define Nn   2048
#define EMBD 256
#define Hh   8
#define HD   32
#define Tt   24
#define Ee   65536

typedef __attribute__((ext_vector_type(16))) __bf16 v16bf;
typedef __attribute__((ext_vector_type(8)))  __bf16 bf16x8;
typedef __attribute__((ext_vector_type(8)))  float  v8f;
typedef __attribute__((ext_vector_type(8)))  float  f32x8;

// Assemble a 16-element bf16 WMMA operand from two contiguous 8-element (16B) loads.
static __device__ inline v16bf load2x8(const __bf16* lo, const __bf16* hi) {
    bf16x8 l = *(const bf16x8*)lo;
    bf16x8 h = *(const bf16x8*)hi;
    v16bf r;
#pragma unroll
    for (int i = 0; i < 8; ++i) { r[i] = l[i]; r[i + 8] = h[i]; }
    return r;
}

static __device__ inline v8f wmma_bf16(v16bf a, v16bf b, v8f c) {
    return __builtin_amdgcn_wmma_f32_16x16x32_bf16(
        /*neg_a=*/false, a, /*neg_b=*/false, b,
        /*c_mod=*/(short)0, c, /*reuse_a=*/false, /*reuse_b=*/false);
}

// ---------------------------------------------------------------- converts ---
__global__ __launch_bounds__(256) void cvt_f32_bf16(const float* __restrict__ src,
                                                    __bf16* __restrict__ dst, int n) {
    int i = blockIdx.x * 256 + threadIdx.x;
    if (i < n) dst[i] = (__bf16)src[i];
}

// temporal_embeddings gather (clip) + convert
__global__ __launch_bounds__(256) void gather_emb(const float* __restrict__ temb,
                                                  const int* __restrict__ tindex,
                                                  __bf16* __restrict__ dst) {
    int i = blockIdx.x * 256 + threadIdx.x;   // over Nn*EMBD
    int nidx = i >> 8, e = i & 255;
    int t = tindex[nidx];
    t = t < 0 ? 0 : (t > Tt - 1 ? Tt - 1 : t);
    dst[i] = (__bf16)temb[t * EMBD + e];
}

// W[k][c] f32 -> Wt[c][k] bf16  (256x256)
__global__ __launch_bounds__(256) void cvt_transpose_w(const float* __restrict__ W,
                                                       __bf16* __restrict__ Wt) {
    int i = blockIdx.x * 256 + threadIdx.x;   // over 65536
    int k = i >> 8, c = i & 255;
    Wt[c * EMBD + k] = (__bf16)W[i];
}

// ------------------------------------------------------------ projections ---
// Y = A[MxK=2048x256] @ W[256x256] + bias, via bf16 WMMA, K in 8 chunks of 32.
// MODE 0: out bf16 [H][N][HD]   (Q/K head layout)
// MODE 1: out bf16 [H][HD][N]   (V transposed, B-operand-friendly)
// MODE 2: out f32  [N][256]     (final output)
template <int MODE>
__global__ __launch_bounds__(32) void gemm_proj(const __bf16* __restrict__ A,
                                                const __bf16* __restrict__ Bt,
                                                const float* __restrict__ bias,
                                                void* __restrict__ out) {
    int lane = threadIdx.x;
    int ml = lane & 15, hi = lane >> 4;
    int row = blockIdx.x * 16 + ml;   // A row for this lane
    int col = blockIdx.y * 16 + ml;   // B column for this lane
    const __bf16* ap = A + (size_t)row * EMBD;
    const __bf16* bp = Bt + (size_t)col * EMBD;
    v8f c = {};
#pragma unroll
    for (int kc = 0; kc < 8; ++kc) {
        v16bf a = load2x8(ap + kc * 32 + hi * 8, ap + kc * 32 + 16 + hi * 8);
        v16bf b = load2x8(bp + kc * 32 + hi * 8, bp + kc * 32 + 16 + hi * 8);
        c = wmma_bf16(a, b, c);
    }
#pragma unroll
    for (int r = 0; r < 8; ++r) {
        int orow = blockIdx.x * 16 + r + 8 * hi;
        int ocol = blockIdx.y * 16 + ml;
        float v = c[r] + bias[ocol];
        if (MODE == 0) {
            int h = ocol >> 5, d = ocol & 31;
            ((__bf16*)out)[((size_t)h * Nn + orow) * HD + d] = (__bf16)v;
        } else if (MODE == 1) {
            int h = ocol >> 5, d = ocol & 31;
            ((__bf16*)out)[((size_t)h * HD + d) * Nn + orow] = (__bf16)v;
        } else {
            ((float*)out)[(size_t)orow * EMBD + ocol] = v;
        }
    }
}

// ----------------------------------------------------------------- scores ---
// S[h,i,j] = scale * q_i . k_j, masked to -inf across graphs. One WMMA/tile.
__global__ __launch_bounds__(32) void scores_kernel(const __bf16* __restrict__ Qb,
                                                    const __bf16* __restrict__ Kb,
                                                    const int* __restrict__ batch,
                                                    float* __restrict__ S, float scale) {
    int lane = threadIdx.x, ml = lane & 15, hi = lane >> 4;
    int h = blockIdx.z;
    int qrow = blockIdx.x * 16 + ml;
    int kcol = blockIdx.y * 16 + ml;
    const __bf16* qp = Qb + ((size_t)h * Nn + qrow) * HD;
    const __bf16* kp = Kb + ((size_t)h * Nn + kcol) * HD;
    v16bf a = load2x8(qp + hi * 8, qp + 16 + hi * 8);     // A: 16x32, k = HD
    v16bf b = load2x8(kp + hi * 8, kp + 16 + hi * 8);     // B: K^T column = key row
    v8f c = {};
    c = wmma_bf16(a, b, c);
    int j = blockIdx.y * 16 + ml;
    int bj = batch[j];
#pragma unroll
    for (int r = 0; r < 8; ++r) {
        int i = blockIdx.x * 16 + r + 8 * hi;
        float s = c[r] * scale;
        if (batch[i] != bj) s = -__builtin_inff();
        S[((size_t)h * Nn + i) * Nn + j] = s;
    }
}

// -------------------------------------------------------------- edge bias ---
// scores[h,src,dst] += edge_attr@We + be  (scatter; -inf mask absorbs adds)
__global__ __launch_bounds__(256) void edge_bias_kernel(const int* __restrict__ eidx,
                                                        const float* __restrict__ eattr,
                                                        const float* __restrict__ We,
                                                        const float* __restrict__ be,
                                                        float* __restrict__ S) {
    int e = blockIdx.x * 256 + threadIdx.x;
    if (e >= Ee) return;
    int src = eidx[e];
    int dst = eidx[Ee + e];
    if (src < 0 || src >= Nn || dst < 0 || dst >= Nn) return;
    float a0 = eattr[e * 4 + 0], a1 = eattr[e * 4 + 1];
    float a2 = eattr[e * 4 + 2], a3 = eattr[e * 4 + 3];
#pragma unroll
    for (int h = 0; h < Hh; ++h) {
        float v = be[h] + a0 * We[0 * Hh + h] + a1 * We[1 * Hh + h]
                        + a2 * We[2 * Hh + h] + a3 * We[3 * Hh + h];
        __hip_atomic_fetch_add(&S[((size_t)h * Nn + src) * Nn + dst], v,
                               __ATOMIC_RELAXED, __HIP_MEMORY_SCOPE_AGENT);
    }
}

// ------------------------------------------------- fused softmax + P@V ------
// One block = (head h, 16 query rows). Two-pass row stats from L2-resident
// scores, then 8 waves split the 2048 keys, P->bf16 in regs, WMMA vs V^T,
// LDS cross-wave reduction, write/accumulate into combined [N][256].
__global__ __launch_bounds__(256) void attn_pv(const float* __restrict__ S,
                                               const __bf16* __restrict__ Vbt,
                                               float* __restrict__ comb,
                                               int accumulate) {
    __shared__ float part[16][16];
    __shared__ float rmax[16], rsum[16];
    __shared__ float outbuf[8][16][32];
    int tid = threadIdx.x;
    int h = blockIdx.y;
    int rowblock = blockIdx.x * 16;
    const float* srow = S + ((size_t)h * Nn + rowblock) * Nn;

    // ---- row stats (max, sum of exp) ----
    int rr = tid >> 4, sub = tid & 15;
    const float* p = srow + (size_t)rr * Nn;
    float mx = -__builtin_inff();
    for (int j = sub; j < Nn; j += 16) mx = fmaxf(mx, p[j]);
    part[rr][sub] = mx;
    __syncthreads();
    if (sub == 0) {
        float m = part[rr][0];
        for (int t = 1; t < 16; ++t) m = fmaxf(m, part[rr][t]);
        rmax[rr] = m;
    }
    __syncthreads();
    float m = rmax[rr];
    float sm = 0.f;
    for (int j = sub; j < Nn; j += 16) sm += __expf(p[j] - m);
    part[rr][sub] = sm;
    __syncthreads();
    if (sub == 0) {
        float s = 0.f;
        for (int t = 0; t < 16; ++t) s += part[rr][t];
        rsum[rr] = s;
    }
    __syncthreads();

    // ---- P @ V via WMMA: wave w covers keys [w*256, w*256+256) ----
    int wave = tid >> 5, lane = tid & 31;
    int ml = lane & 15, hi = lane >> 4;
    float mrow = rmax[ml];
    const float* arow = srow + (size_t)ml * Nn;
    const __bf16* v0p = Vbt + ((size_t)h * HD + ml) * Nn;        // d = ml
    const __bf16* v1p = Vbt + ((size_t)h * HD + 16 + ml) * Nn;   // d = 16+ml
    v8f c0 = {}, c1 = {};
    int keybase = wave * 256;
#pragma unroll 1
    for (int kc = 0; kc < 8; ++kc) {
        int k0 = keybase + kc * 32;
        f32x8 slo = *(const f32x8*)&arow[k0 + hi * 8];
        f32x8 shi = *(const f32x8*)&arow[k0 + 16 + hi * 8];
        v16bf a;
#pragma unroll
        for (int i = 0; i < 8; ++i) {
            a[i]     = (__bf16)__expf(slo[i] - mrow);   // exp(-inf)=0: mask ok
            a[i + 8] = (__bf16)__expf(shi[i] - mrow);
        }
        v16bf b0 = load2x8(v0p + k0 + hi * 8, v0p + k0 + 16 + hi * 8);
        v16bf b1 = load2x8(v1p + k0 + hi * 8, v1p + k0 + 16 + hi * 8);
        c0 = wmma_bf16(a, b0, c0);
        c1 = wmma_bf16(a, b1, c1);
    }
#pragma unroll
    for (int r = 0; r < 8; ++r) {
        int orow = r + 8 * hi;
        outbuf[wave][orow][ml] = c0[r];
        outbuf[wave][orow][16 + ml] = c1[r];
    }
    __syncthreads();
    for (int e = tid; e < 512; e += 256) {
        int row = e >> 5, d = e & 31;
        float s = 0.f;
#pragma unroll
        for (int w = 0; w < 8; ++w) s += outbuf[w][row][d];
        s /= rsum[row];
        size_t o = (size_t)(rowblock + row) * EMBD + h * HD + d;
        if (accumulate) comb[o] += s; else comb[o] = s;
    }
}

// ---------------------------------------------------------------- launch ----
extern "C" void kernel_launch(void* const* d_in, const int* in_sizes, int n_in,
                              void* d_out, int out_size, void* d_ws, size_t ws_size,
                              hipStream_t stream) {
    (void)in_sizes; (void)n_in; (void)out_size; (void)ws_size;
    const float* x      = (const float*)d_in[0];
    const int*   eidx   = (const int*)d_in[1];
    const float* eattr  = (const float*)d_in[2];
    const int*   batch  = (const int*)d_in[3];
    const int*   tindex = (const int*)d_in[4];
    const float* temb   = (const float*)d_in[5];
    const float* Wq  = (const float*)d_in[6];  const float* bq  = (const float*)d_in[7];
    const float* Wk  = (const float*)d_in[8];  const float* bk  = (const float*)d_in[9];
    const float* Wv  = (const float*)d_in[10]; const float* bv  = (const float*)d_in[11];
    const float* Wo  = (const float*)d_in[12]; const float* bo  = (const float*)d_in[13];
    const float* We  = (const float*)d_in[14]; const float* be  = (const float*)d_in[15];
    const float* Wtq = (const float*)d_in[16]; const float* btq = (const float*)d_in[17];
    const float* Wtk = (const float*)d_in[18]; const float* btk = (const float*)d_in[19];
    const float* Wtv = (const float*)d_in[20]; const float* btv = (const float*)d_in[21];

    char* ws = (char*)d_ws;
    size_t off = 0;
    auto alloc = [&](size_t bytes) -> char* {
        char* ptr = ws + off;
        off += (bytes + 255) & ~(size_t)255;
        return ptr;
    };
    __bf16* xb   = (__bf16*)alloc((size_t)Nn * EMBD * 2);
    __bf16* embb = (__bf16*)alloc((size_t)Nn * EMBD * 2);
    __bf16* Wqt  = (__bf16*)alloc((size_t)EMBD * EMBD * 2);
    __bf16* Wkt  = (__bf16*)alloc((size_t)EMBD * EMBD * 2);
    __bf16* Wvt  = (__bf16*)alloc((size_t)EMBD * EMBD * 2);
    __bf16* Wtqt = (__bf16*)alloc((size_t)EMBD * EMBD * 2);
    __bf16* Wtkt = (__bf16*)alloc((size_t)EMBD * EMBD * 2);
    __bf16* Wtvt = (__bf16*)alloc((size_t)EMBD * EMBD * 2);
    __bf16* Wot  = (__bf16*)alloc((size_t)EMBD * EMBD * 2);
    __bf16* Qb   = (__bf16*)alloc((size_t)Hh * Nn * HD * 2);
    __bf16* Kb   = (__bf16*)alloc((size_t)Hh * Nn * HD * 2);
    __bf16* Vbt  = (__bf16*)alloc((size_t)Hh * HD * Nn * 2);
    __bf16* TQb  = (__bf16*)alloc((size_t)Hh * Nn * HD * 2);
    __bf16* TKb  = (__bf16*)alloc((size_t)Hh * Nn * HD * 2);
    __bf16* TVbt = (__bf16*)alloc((size_t)Hh * HD * Nn * 2);
    float*  comb = (float*)alloc((size_t)Nn * EMBD * 4);
    __bf16* cb   = (__bf16*)alloc((size_t)Nn * EMBD * 2);
    float*  scor = (float*)alloc((size_t)Hh * Nn * Nn * 4);   // 128 MB, L2-resident

    const float scale = 0.17677669529663687f;   // HD^-0.5

    // input conversions
    cvt_f32_bf16<<<(Nn * EMBD) / 256, 256, 0, stream>>>(x, xb, Nn * EMBD);
    gather_emb<<<(Nn * EMBD) / 256, 256, 0, stream>>>(temb, tindex, embb);
    cvt_transpose_w<<<256, 256, 0, stream>>>(Wq, Wqt);
    cvt_transpose_w<<<256, 256, 0, stream>>>(Wk, Wkt);
    cvt_transpose_w<<<256, 256, 0, stream>>>(Wv, Wvt);
    cvt_transpose_w<<<256, 256, 0, stream>>>(Wtq, Wtqt);
    cvt_transpose_w<<<256, 256, 0, stream>>>(Wtk, Wtkt);
    cvt_transpose_w<<<256, 256, 0, stream>>>(Wtv, Wtvt);
    cvt_transpose_w<<<256, 256, 0, stream>>>(Wo, Wot);

    // projections (WMMA)
    dim3 ggrid(Nn / 16, EMBD / 16);
    gemm_proj<0><<<ggrid, 32, 0, stream>>>(xb, Wqt, bq, Qb);
    gemm_proj<0><<<ggrid, 32, 0, stream>>>(xb, Wkt, bk, Kb);
    gemm_proj<1><<<ggrid, 32, 0, stream>>>(xb, Wvt, bv, Vbt);
    gemm_proj<0><<<ggrid, 32, 0, stream>>>(embb, Wtqt, btq, TQb);
    gemm_proj<0><<<ggrid, 32, 0, stream>>>(embb, Wtkt, btk, TKb);
    gemm_proj<1><<<ggrid, 32, 0, stream>>>(embb, Wtvt, btv, TVbt);

    // spatial attention
    dim3 sgrid(Nn / 16, Nn / 16, Hh);
    scores_kernel<<<sgrid, 32, 0, stream>>>(Qb, Kb, batch, scor, scale);
    edge_bias_kernel<<<Ee / 256, 256, 0, stream>>>(eidx, eattr, We, be, scor);
    attn_pv<<<dim3(Nn / 16, Hh), 256, 0, stream>>>(scor, Vbt, comb, 0);

    // temporal attention (reuse scores buffer)
    scores_kernel<<<sgrid, 32, 0, stream>>>(TQb, TKb, batch, scor, scale);
    attn_pv<<<dim3(Nn / 16, Hh), 256, 0, stream>>>(scor, TVbt, comb, 1);

    // output projection
    cvt_f32_bf16<<<(Nn * EMBD) / 256, 256, 0, stream>>>(comb, cb, Nn * EMBD);
    gemm_proj<2><<<ggrid, 32, 0, stream>>>(cb, Wot, bo, d_out);
}
